// PointTransformerModel_56427280335319
// MI455X (gfx1250) — compile-verified, matrix operand order checked by reading, not run on hardware
//
#include <hip/hip_runtime.h>
#include <math.h>

// ---------------------------------------------------------------------------
// Point Transformer forward for MI455X (gfx1250, wave32, WMMA).
// Dense layers: WMMA f16->f32 GEMM (v_wmma_f32_16x16x32_f16). Weights are
// pre-packed once to transposed K-padded f16 so B fragments are contiguous
// b128 loads. A fragments are b128 f32 loads + v_cvt_pk_f16_f32.
//   k_gemm_f4: 16x64 strip per wave (A reused across 4 WMMAs) - dominant path
//   k_gemm_f : 16x16 tile per wave (N=16/32)
//   k_gemm_g : ragged shapes, clamp-address + select-value guards
// ---------------------------------------------------------------------------

typedef __attribute__((ext_vector_type(16))) _Float16 v16h;
typedef __attribute__((ext_vector_type(8)))  _Float16 v8h;
typedef __attribute__((ext_vector_type(8)))  float    v8f;
typedef __attribute__((ext_vector_type(4)))  float    v4f;

#define BB 8          // batch
#define NN0 4096      // initial points
#define KNN 16        // neighbors

// ---------------------------------------------------------------------------
// Weight pre-pack: W[K,N] f32 (row-major)  ->  Wt[N,Kp] f16, zero-padded K.
// ---------------------------------------------------------------------------
__global__ void k_pack_w(const float* __restrict__ W, _Float16* __restrict__ Wt,
                         int K, int N, int Kp)
{
    int e = blockIdx.x * blockDim.x + threadIdx.x;   // over N*Kp
    if (e >= N * Kp) return;
    int n = e / Kp, k = e - n * Kp;
    int kc = (k < K) ? k : (K - 1);
    float v = W[(size_t)kc * N + n];
    Wt[e] = (_Float16)((k < K) ? v : 0.f);
}

__device__ __forceinline__ v16h load_afrag(const float* ap, int kk)
{
    v4f a0 = *(const v4f*)(ap + kk);
    v4f a1 = *(const v4f*)(ap + kk + 4);
    v4f a2 = *(const v4f*)(ap + kk + 16);
    v4f a3 = *(const v4f*)(ap + kk + 20);
    v16h a;
#pragma unroll
    for (int i = 0; i < 4; ++i) {
        a[i]      = (_Float16)a0[i];
        a[4 + i]  = (_Float16)a1[i];
        a[8 + i]  = (_Float16)a2[i];
        a[12 + i] = (_Float16)a3[i];
    }
    return a;
}

__device__ __forceinline__ v16h load_bfrag(const _Float16* bp, int kk)
{
    v8h bl = *(const v8h*)(bp + kk);
    v8h bh = *(const v8h*)(bp + kk + 16);
    return __builtin_shufflevector(bl, bh, 0, 1, 2, 3, 4, 5, 6, 7,
                                   8, 9, 10, 11, 12, 13, 14, 15);
}

// ---------------------------------------------------------------------------
// Strip GEMM: 16 rows x 64 cols per wave; A fragment reused by 4 WMMAs.
// All 4 B fragments loaded up front so waits stagger across the WMMAs.
// Requires M%16==0, N%64==0, K%32==0.
// ---------------------------------------------------------------------------
__global__ __launch_bounds__(32) void k_gemm_f4(
    const float* __restrict__ A, const _Float16* __restrict__ Wt,
    const float* __restrict__ bias, const float* __restrict__ res,
    float* __restrict__ C, int M, int K, int N, int relu)
{
    const int lane = threadIdx.x;
    const int tm = blockIdx.x * 16;
    const int tn = blockIdx.y * 64;
    const int grp = lane >> 4;
    const int l15 = lane & 15;

    const float*    ap = A  + (size_t)(tm + l15) * K + grp * 8;
    const _Float16* bp = Wt + (size_t)(tn + l15) * K + grp * 8;
    const size_t bs = (size_t)16 * K;                 // 16 columns of packed W

    v8f acc[4] = {{}, {}, {}, {}};
    for (int kk = 0; kk < K; kk += 32) {
        v16h a  = load_afrag(ap, kk);
        v16h b0 = load_bfrag(bp,          kk);
        v16h b1 = load_bfrag(bp + bs,     kk);
        v16h b2 = load_bfrag(bp + 2 * bs, kk);
        v16h b3 = load_bfrag(bp + 3 * bs, kk);
        acc[0] = __builtin_amdgcn_wmma_f32_16x16x32_f16(false, a, false, b0,
                                                        (short)0, acc[0], false, false);
        acc[1] = __builtin_amdgcn_wmma_f32_16x16x32_f16(false, a, false, b1,
                                                        (short)0, acc[1], false, false);
        acc[2] = __builtin_amdgcn_wmma_f32_16x16x32_f16(false, a, false, b2,
                                                        (short)0, acc[2], false, false);
        acc[3] = __builtin_amdgcn_wmma_f32_16x16x32_f16(false, a, false, b3,
                                                        (short)0, acc[3], false, false);
    }

    const int mbase = tm + grp * 8;
#pragma unroll
    for (int nt = 0; nt < 4; ++nt) {
        const int col = tn + nt * 16 + l15;
        const float bv = bias ? bias[col] : 0.f;
#pragma unroll
        for (int r = 0; r < 8; ++r) {
            int row = mbase + r;
            float v = acc[nt][r] + bv;
            if (res)  v += res[(size_t)row * N + col];
            if (relu) v = fmaxf(v, 0.f);
            C[(size_t)row * N + col] = v;
        }
    }
}

// ---------------------------------------------------------------------------
// Fast GEMM: 16x16 tile per wave.  Requires M%16==0, N%16==0, K%32==0.
// ---------------------------------------------------------------------------
__global__ __launch_bounds__(32) void k_gemm_f(
    const float* __restrict__ A, const _Float16* __restrict__ Wt,
    const float* __restrict__ bias, const float* __restrict__ res,
    float* __restrict__ C, int M, int K, int N, int relu)
{
    const int lane = threadIdx.x;
    const int tm = blockIdx.x * 16;
    const int tn = blockIdx.y * 16;
    const int grp = lane >> 4;
    const int l15 = lane & 15;

    const float*    ap = A  + (size_t)(tm + l15) * K + grp * 8;
    const _Float16* bp = Wt + (size_t)(tn + l15) * K + grp * 8;

    v8f acc = {};
#pragma unroll 2
    for (int kk = 0; kk < K; kk += 32) {
        v16h a = load_afrag(ap, kk);
        v16h b = load_bfrag(bp, kk);
        acc = __builtin_amdgcn_wmma_f32_16x16x32_f16(false, a, false, b,
                                                     (short)0, acc, false, false);
    }

    const int col = tn + l15;
    const int mbase = tm + grp * 8;
    const float bv = bias ? bias[col] : 0.f;
#pragma unroll
    for (int r = 0; r < 8; ++r) {
        int row = mbase + r;
        float v = acc[r] + bv;
        if (res)  v += res[(size_t)row * N + col];
        if (relu) v = fmaxf(v, 0.f);
        C[(size_t)row * N + col] = v;
    }
}

// ---------------------------------------------------------------------------
// Generic GEMM: arbitrary M, K, N.  Addresses clamped in-range, values
// selected to zero past K; stores guarded.  B is K-padded.
// ---------------------------------------------------------------------------
__global__ __launch_bounds__(32) void k_gemm_g(
    const float* __restrict__ A, const _Float16* __restrict__ Wt,
    const float* __restrict__ bias, const float* __restrict__ res,
    float* __restrict__ C, int M, int K, int Kp, int N, int relu)
{
    const int lane = threadIdx.x;
    const int tm = blockIdx.x * 16;
    const int tn = blockIdx.y * 16;
    const int grp = lane >> 4;
    const int l15 = lane & 15;

    const int arow = tm + l15;
    const int bcol = tn + l15;
    const int ar = (arow < M) ? arow : (M - 1);
    const int bc = (bcol < N) ? bcol : (N - 1);

    const float*    ap = A  + (size_t)ar * K;
    const _Float16* bp = Wt + (size_t)bc * Kp + grp * 8;

    v8f acc = {};
    for (int kk = 0; kk < Kp; kk += 32) {
        const int kb = kk + grp * 8;
        v16h a, b;
#pragma unroll
        for (int i = 0; i < 8; ++i) {
            int k0 = kb + i;
            int k1 = kb + 16 + i;
            float av0 = ap[(k0 < K) ? k0 : (K - 1)];
            float av1 = ap[(k1 < K) ? k1 : (K - 1)];
            a[i]     = (_Float16)((k0 < K) ? av0 : 0.f);
            a[8 + i] = (_Float16)((k1 < K) ? av1 : 0.f);
            b[i]     = bp[kk + i];
            b[8 + i] = bp[kk + 16 + i];
        }
        acc = __builtin_amdgcn_wmma_f32_16x16x32_f16(false, a, false, b,
                                                     (short)0, acc, false, false);
    }

    const int col = tn + l15;
    const int mbase = tm + grp * 8;
    if (col < N) {
        const float bv = bias ? bias[col] : 0.f;
#pragma unroll
        for (int r = 0; r < 8; ++r) {
            int row = mbase + r;
            if (row < M) {
                float v = acc[r] + bv;
                if (res)  v += res[(size_t)row * N + col];
                if (relu) v = fmaxf(v, 0.f);
                C[(size_t)row * N + col] = v;
            }
        }
    }
}

// --------------------------- xyz = x[..., :3] -------------------------------
__global__ void k_xyz_extract(const float* __restrict__ x, float* __restrict__ xyz,
                              int total)
{
    int i = blockIdx.x * blockDim.x + threadIdx.x;
    if (i >= total) return;
    xyz[i * 3 + 0] = x[i * 6 + 0];
    xyz[i * 3 + 1] = x[i * 6 + 1];
    xyz[i * 3 + 2] = x[i * 6 + 2];
}

// --------------------------- KNN (top-16 smallest) --------------------------
__global__ void k_knn(const float* __restrict__ q, const float* __restrict__ r,
                      int* __restrict__ idx, int Mq, int Nr)
{
    int m = blockIdx.x * blockDim.x + threadIdx.x;
    int b = blockIdx.y;
    if (m >= Mq) return;
    const float* qp = q + ((size_t)b * Mq + m) * 3;
    const float* rp = r + (size_t)b * Nr * 3;
    float qx = qp[0], qy = qp[1], qz = qp[2];
    float bd[KNN]; int bi[KNN];
#pragma unroll
    for (int j = 0; j < KNN; ++j) { bd[j] = 3.4e38f; bi[j] = 0; }
    for (int n = 0; n < Nr; ++n) {
        float dx = qx - rp[n * 3 + 0];
        float dy = qy - rp[n * 3 + 1];
        float dz = qz - rp[n * 3 + 2];
        float d = dx * dx + dy * dy + dz * dz;
        if (d < bd[KNN - 1]) {
            float cd = d; int ci = n;
#pragma unroll
            for (int j = 0; j < KNN; ++j) {
                if (cd < bd[j]) {
                    float td = bd[j]; int ti = bi[j];
                    bd[j] = cd; bi[j] = ci; cd = td; ci = ti;
                }
            }
        }
    }
    int* op = idx + ((size_t)b * Mq + m) * KNN;
#pragma unroll
    for (int j = 0; j < KNN; ++j) op[j] = bi[j];
}

// --------------------------- FPS (one block per batch) ----------------------
#define FPS_T 256
__global__ void k_fps(const float* __restrict__ xyz, int* __restrict__ sidx,
                      int N, int npoint)
{
    __shared__ float dist[NN0];
    __shared__ float rd[FPS_T];
    __shared__ int   ri[FPS_T];
    __shared__ int   s_last;
    int b = blockIdx.x, t = threadIdx.x;
    const float* xp = xyz + (size_t)b * N * 3;
    for (int i = t; i < N; i += FPS_T) dist[i] = 1e10f;
    if (t == 0) { sidx[b * npoint] = 0; s_last = 0; }
    __syncthreads();
    for (int s = 1; s < npoint; ++s) {
        int last = s_last;
        float lx = xp[last * 3 + 0], ly = xp[last * 3 + 1], lz = xp[last * 3 + 2];
        float bestd = -1.f; int besti = 0x7fffffff;
        for (int i = t; i < N; i += FPS_T) {
            float dx = xp[i * 3 + 0] - lx;
            float dy = xp[i * 3 + 1] - ly;
            float dz = xp[i * 3 + 2] - lz;
            float d = dx * dx + dy * dy + dz * dz;
            float nd = fminf(dist[i], d);
            dist[i] = nd;
            if (nd > bestd) { bestd = nd; besti = i; }
        }
        rd[t] = bestd; ri[t] = besti;
        __syncthreads();
        for (int off = FPS_T / 2; off > 0; off >>= 1) {
            if (t < off) {
                if (rd[t + off] > rd[t] ||
                    (rd[t + off] == rd[t] && ri[t + off] < ri[t])) {
                    rd[t] = rd[t + off]; ri[t] = ri[t + off];
                }
            }
            __syncthreads();
        }
        if (t == 0) { s_last = ri[0]; sidx[b * npoint + s] = ri[0]; }
        __syncthreads();
    }
}

// --------------------------- gather new_xyz = xyz[sidx] ---------------------
__global__ void k_gather_xyz(const float* __restrict__ xyz, const int* __restrict__ sidx,
                             float* __restrict__ out, int N, int Npn)
{
    int m = blockIdx.x * blockDim.x + threadIdx.x;
    int b = blockIdx.y;
    if (m >= Npn) return;
    int id = sidx[b * Npn + m];
    const float* s = xyz + ((size_t)b * N + id) * 3;
    float* d = out + ((size_t)b * Npn + m) * 3;
    d[0] = s[0]; d[1] = s[1]; d[2] = s[2];
}

// ------------- g = concat(xyz[idx]-new_xyz, feats[idx]) ---------------------
__global__ void k_groupcat(const float* __restrict__ xyz, const float* __restrict__ nxyz,
                           const float* __restrict__ feats, const int* __restrict__ idx,
                           float* __restrict__ g, int Npn, int N, int dp)
{
    int e = blockIdx.x * blockDim.x + threadIdx.x;
    int total = BB * Npn * KNN;
    if (e >= total) return;
    int b = e / (Npn * KNN);
    int rme = e - b * Npn * KNN;
    int m = rme >> 4;
    int id = idx[e];
    const float* src = xyz  + ((size_t)b * N   + id) * 3;
    const float* nc  = nxyz + ((size_t)b * Npn + m) * 3;
    float* gp = g + (size_t)e * (3 + dp);
    gp[0] = src[0] - nc[0];
    gp[1] = src[1] - nc[1];
    gp[2] = src[2] - nc[2];
    const float* fp = feats + ((size_t)b * N + id) * dp;
    for (int c = 0; c < dp; ++c) gp[3 + c] = fp[c];
}

// --------------------------- max over neighbors -----------------------------
__global__ void k_maxpool(const float* __restrict__ h, float* __restrict__ out,
                          int rows, int d)
{
    int e = blockIdx.x * blockDim.x + threadIdx.x;
    if (e >= rows * d) return;
    int r = e / d, c = e - r * d;
    const float* hp = h + ((size_t)r * KNN) * d + c;
    float v = hp[0];
#pragma unroll
    for (int j = 1; j < KNN; ++j) v = fmaxf(v, hp[(size_t)j * d]);
    out[e] = v;
}

// -------------- posin = xyz[m] - xyz[idx[m,j]]  (per batch) -----------------
__global__ void k_posin(const float* __restrict__ xyz, const int* __restrict__ idx,
                        float* __restrict__ out, int Np)
{
    int e = blockIdx.x * blockDim.x + threadIdx.x;
    if (e >= Np * KNN) return;
    int m = e >> 4;
    int id = idx[e];
    out[e * 3 + 0] = xyz[m * 3 + 0] - xyz[id * 3 + 0];
    out[e * 3 + 1] = xyz[m * 3 + 1] - xyz[id * 3 + 1];
    out[e * 3 + 2] = xyz[m * 3 + 2] - xyz[id * 3 + 2];
}

// -------------- attin = q[m] - xk[idx] + pos  (per batch) -------------------
__global__ void k_attin(const float* __restrict__ q, const float* __restrict__ xk,
                        const float* __restrict__ pos, const int* __restrict__ idx,
                        float* __restrict__ out, int Np)
{
    int e = blockIdx.x * blockDim.x + threadIdx.x;
    if (e >= Np * KNN * 64) return;
    int mj = e >> 6;
    int c  = e & 63;
    int m  = mj >> 4;
    int id = idx[mj];
    out[e] = q[m * 64 + c] - xk[id * 64 + c] + pos[e];
}

// ---- res[m,c] = sum_j softmax_j(logit/8)[j] * (xv[idx,c] + pos[m,j,c]) -----
__global__ void k_smred(const float* __restrict__ logit, const float* __restrict__ pos,
                        const float* __restrict__ xv, const int* __restrict__ idx,
                        float* __restrict__ res, int Np)
{
    int e = blockIdx.x * blockDim.x + threadIdx.x;
    if (e >= Np * 64) return;
    int m = e >> 6;
    int c = e & 63;
    const float sc = 0.125f;                            // 1/sqrt(64)
    float lg[KNN];
    float mx = -3.4e38f;
#pragma unroll
    for (int j = 0; j < KNN; ++j) {
        lg[j] = logit[(((size_t)m * KNN + j) << 6) + c] * sc;
        mx = fmaxf(mx, lg[j]);
    }
    float se = 0.f;
#pragma unroll
    for (int j = 0; j < KNN; ++j) { lg[j] = __expf(lg[j] - mx); se += lg[j]; }
    float inv = 1.f / se, acc = 0.f;
#pragma unroll
    for (int j = 0; j < KNN; ++j) {
        int id = idx[m * KNN + j];
        acc += lg[j] * inv * (xv[(size_t)id * 64 + c] + pos[(((size_t)m * KNN + j) << 6) + c]);
    }
    res[e] = acc;
}

// --------------------------- mean over points -------------------------------
__global__ void k_mean(const float* __restrict__ f, float* __restrict__ g,
                       int Np, int d)
{
    int e = blockIdx.x * blockDim.x + threadIdx.x;
    if (e >= BB * d) return;
    int b = e / d, c = e - b * d;
    float s = 0.f;
    for (int j = 0; j < Np; ++j) s += f[((size_t)b * Np + j) * d + c];
    g[e] = s / (float)Np;
}

// ===========================================================================
// Host orchestration
// ===========================================================================
struct PW { const _Float16* w; int K; int Kp; int N; };

static PW pack(hipStream_t s, const float* W, int K, int N, char*& wp)
{
    int Kp = (K + 31) & ~31;
    _Float16* dst = (_Float16*)wp;
    wp += (((size_t)N * Kp * sizeof(_Float16)) + 255) & ~(size_t)255;
    int tot = N * Kp;
    k_pack_w<<<dim3((tot + 255) / 256), dim3(256), 0, s>>>(W, dst, K, N, Kp);
    PW r; r.w = dst; r.K = K; r.Kp = Kp; r.N = N;
    return r;
}

static inline void gemm(hipStream_t s, const float* A, const PW& w,
                        const float* bias, const float* res, float* C,
                        int M, int relu)
{
    if ((M & 15) == 0 && (w.K & 31) == 0 && (w.N & 63) == 0) {
        dim3 grid(M / 16, w.N / 64);
        k_gemm_f4<<<grid, dim3(32), 0, s>>>(A, w.w, bias, res, C, M, w.K, w.N, relu);
    } else if ((M & 15) == 0 && (w.K & 31) == 0 && (w.N & 15) == 0) {
        dim3 grid(M / 16, w.N / 16);
        k_gemm_f<<<grid, dim3(32), 0, s>>>(A, w.w, bias, res, C, M, w.K, w.N, relu);
    } else {
        dim3 grid((M + 15) / 16, (w.N + 15) / 16);
        k_gemm_g<<<grid, dim3(32), 0, s>>>(A, w.w, bias, res, C, M, w.K, w.Kp, w.N, relu);
    }
}

struct PTp {
    PW d1, d2, f1, f2, g1, g2, wk, wq, wv;
    const float *d1b, *d2b, *f1b, *f2b, *g1b, *g2b;
};

static PTp mkpt(hipStream_t s, void* const* din, int base, int dp, char*& wp)
{
    PTp p;
    p.d1 = pack(s, (const float*)din[base + 0],  3, 64, wp);  p.d1b = (const float*)din[base + 1];
    p.d2 = pack(s, (const float*)din[base + 2], 64, 64, wp);  p.d2b = (const float*)din[base + 3];
    p.f1 = pack(s, (const float*)din[base + 4], dp, 64, wp);  p.f1b = (const float*)din[base + 5];
    p.f2 = pack(s, (const float*)din[base + 6], 64, dp, wp);  p.f2b = (const float*)din[base + 7];
    p.g1 = pack(s, (const float*)din[base + 8], 64, 64, wp);  p.g1b = (const float*)din[base + 9];
    p.g2 = pack(s, (const float*)din[base + 10], 64, 64, wp); p.g2b = (const float*)din[base + 11];
    p.wk = pack(s, (const float*)din[base + 12], 64, 64, wp);
    p.wq = pack(s, (const float*)din[base + 13], 64, 64, wp);
    p.wv = pack(s, (const float*)din[base + 14], 64, 64, wp);
    return p;
}

struct WS {
    float *xyzA, *xyzB, *fA, *fB;
    int   *idx, *sidx;
    float *xb, *qb, *kb, *vb, *rb;
    float *posin, *posb, *t1, *t2;
    float *gbuf, *hbuf, *gm;
};

static void pt_layer(hipStream_t s, const PTp& P, const float* xyz,
                     const float* f, float* fout, int Np, int dp, WS& W)
{
    dim3 gk((Np + 63) / 64, BB);
    k_knn<<<gk, dim3(64), 0, s>>>(xyz, xyz, W.idx, Np, Np);

    gemm(s, f,    P.f1, P.f1b, nullptr, W.xb, BB * Np, 0);
    gemm(s, W.xb, P.wq, nullptr, nullptr, W.qb, BB * Np, 0);
    gemm(s, W.xb, P.wk, nullptr, nullptr, W.kb, BB * Np, 0);
    gemm(s, W.xb, P.wv, nullptr, nullptr, W.vb, BB * Np, 0);

    for (int b = 0; b < BB; ++b) {
        const float* xyzb = xyz + (size_t)b * Np * 3;
        const int*   idxb = W.idx + (size_t)b * Np * KNN;
        const int e1 = Np * KNN;
        k_posin<<<dim3((e1 + 255) / 256), dim3(256), 0, s>>>(xyzb, idxb, W.posin, Np);
        gemm(s, W.posin, P.d1, P.d1b, nullptr, W.t1,   e1, 1);
        gemm(s, W.t1,    P.d2, P.d2b, nullptr, W.posb, e1, 0);
        const int e2 = Np * KNN * 64;
        k_attin<<<dim3((e2 + 255) / 256), dim3(256), 0, s>>>(
            W.qb + (size_t)b * Np * 64, W.kb + (size_t)b * Np * 64,
            W.posb, idxb, W.t1, Np);
        gemm(s, W.t1, P.g1, P.g1b, nullptr, W.t2, e1, 1);
        gemm(s, W.t2, P.g2, P.g2b, nullptr, W.t1, e1, 0);
        const int e3 = Np * 64;
        k_smred<<<dim3((e3 + 255) / 256), dim3(256), 0, s>>>(
            W.t1, W.posb, W.vb + (size_t)b * Np * 64, idxb,
            W.rb + (size_t)b * Np * 64, Np);
    }
    gemm(s, W.rb, P.f2, P.f2b, f, fout, BB * Np, 0);   // fc2 + residual
}

static void tdown(hipStream_t s, const PW& dw, const float* db,
                  const float* xyz, float* xyznew, const float* f, float* fnew,
                  int N, int Npn, int dpin, int dpout, WS& W)
{
    k_fps<<<dim3(BB), dim3(FPS_T), 0, s>>>(xyz, W.sidx, N, Npn);
    k_gather_xyz<<<dim3((Npn + 63) / 64, BB), dim3(64), 0, s>>>(xyz, W.sidx, xyznew, N, Npn);
    k_knn<<<dim3((Npn + 63) / 64, BB), dim3(64), 0, s>>>(xyznew, xyz, W.idx, Npn, N);
    const int tot = BB * Npn * KNN;
    k_groupcat<<<dim3((tot + 255) / 256), dim3(256), 0, s>>>(
        xyz, xyznew, f, W.idx, W.gbuf, Npn, N, dpin);
    gemm(s, W.gbuf, dw, db, nullptr, W.hbuf, tot, 1);
    const int tot2 = BB * Npn * dpout;
    k_maxpool<<<dim3((tot2 + 255) / 256), dim3(256), 0, s>>>(W.hbuf, fnew, BB * Npn, dpout);
    (void)dpout;
}

extern "C" void kernel_launch(void* const* d_in, const int* in_sizes, int n_in,
                              void* d_out, int out_size, void* d_ws, size_t ws_size,
                              hipStream_t stream)
{
    (void)in_sizes; (void)n_in; (void)out_size; (void)ws_size;

    const float* x = (const float*)d_in[0];
    char* wp = (char*)d_ws;

    // ---- pre-pack all weights (pytree order: down, mlp1_*, mlp2_*, t[], t0) ----
    const int dnin[4]  = {35, 67, 131, 259};
    const int dnout[4] = {64, 128, 256, 512};
    PW downW[4]; const float* downB[4];
    for (int i = 0; i < 4; ++i) {
        downW[i] = pack(stream, (const float*)d_in[1 + 2 * i], dnin[i], dnout[i], wp);
        downB[i] = (const float*)d_in[2 + 2 * i];
    }
    PW m11 = pack(stream, (const float*)d_in[9],  6,   16,  wp);
    const float* m11b = (const float*)d_in[10];
    PW m12 = pack(stream, (const float*)d_in[11], 16,  32,  wp);
    const float* m12b = (const float*)d_in[12];
    PW m21 = pack(stream, (const float*)d_in[13], 512, 256, wp);
    const float* m21b = (const float*)d_in[14];
    PW m22 = pack(stream, (const float*)d_in[15], 256, 128, wp);
    const float* m22b = (const float*)d_in[16];
    PW m23 = pack(stream, (const float*)d_in[17], 128, 40,  wp);
    const float* m23b = (const float*)d_in[18];
    const int dpts[4] = {64, 128, 256, 512};
    PTp pt[4];
    for (int i = 0; i < 4; ++i) pt[i] = mkpt(stream, d_in, 19 + 15 * i, dpts[i], wp);
    PTp pt0 = mkpt(stream, d_in, 79, 32, wp);

    // ---- carve float workspace ----
    auto af = [&](size_t nflt) -> float* {
        float* r = (float*)wp;
        wp += ((nflt * sizeof(float)) + 255) & ~(size_t)255;
        return r;
    };
    WS W;
    W.xyzA = af((size_t)BB * NN0 * 3);
    W.xyzB = af((size_t)BB * NN0 * 3);
    W.fA   = af((size_t)BB * NN0 * 32);
    W.fB   = af((size_t)BB * NN0 * 32);
    W.idx  = (int*)af((size_t)BB * NN0 * KNN);
    W.sidx = (int*)af((size_t)BB * 1024);
    W.xb   = af((size_t)BB * NN0 * 64);
    W.qb   = af((size_t)BB * NN0 * 64);
    W.kb   = af((size_t)BB * NN0 * 64);
    W.vb   = af((size_t)BB * NN0 * 64);
    W.rb   = af((size_t)BB * NN0 * 64);
    W.posin= af((size_t)NN0 * KNN * 3);
    W.posb = af((size_t)NN0 * KNN * 64);
    W.t1   = af((size_t)NN0 * KNN * 64);
    W.t2   = af((size_t)NN0 * KNN * 64);
    W.gbuf = af((size_t)BB * 1024 * KNN * 35);
    W.hbuf = af((size_t)BB * 1024 * KNN * 64);
    W.gm   = af((size_t)BB * 512);

    const int total0 = BB * NN0;

    k_xyz_extract<<<dim3((total0 + 255) / 256), dim3(256), 0, stream>>>(x, W.xyzA, total0);

    // f = dense(relu(dense(x, mlp1_1)), mlp1_2)
    gemm(stream, x,    m11, m11b, nullptr, W.t1, total0, 1);
    gemm(stream, W.t1, m12, m12b, nullptr, W.fA, total0, 0);

    float* xyzCur = W.xyzA; float* xyzAlt = W.xyzB;
    float* fCur   = W.fA;   float* fAlt   = W.fB;

    pt_layer(stream, pt0, xyzCur, fCur, fAlt, NN0, 32, W);
    { float* t = fCur; fCur = fAlt; fAlt = t; }

    const int npl[4] = {1024, 256, 64, 16};
    int Np = NN0, dpin = 32;
    for (int i = 0; i < 4; ++i) {
        tdown(stream, downW[i], downB[i], xyzCur, xyzAlt, fCur, fAlt,
              Np, npl[i], dpin, dpts[i], W);
        { float* t = xyzCur; xyzCur = xyzAlt; xyzAlt = t; }
        { float* t = fCur;   fCur   = fAlt;   fAlt   = t; }
        Np = npl[i]; dpin = dpts[i];
        pt_layer(stream, pt[i], xyzCur, fCur, fAlt, Np, dpin, W);
        { float* t = fCur; fCur = fAlt; fAlt = t; }
    }

    // head: mean over 16 points -> 512 -> 256 -> 128 -> 40
    k_mean<<<dim3((BB * 512 + 255) / 256), dim3(256), 0, stream>>>(fCur, W.gm, Np, 512);
    gemm(stream, W.gm, m21, m21b, nullptr, W.t1, BB, 1);
    gemm(stream, W.t1, m22, m22b, nullptr, W.t2, BB, 1);
    gemm(stream, W.t2, m23, m23b, nullptr, (float*)d_out, BB, 0);
}
